// EncoderProcesserDecoder_59296318488579
// MI455X (gfx1250) — compile-verified, matrix operand order checked by reading, not run on hardware
//
#include <hip/hip_runtime.h>
#include <hip/hip_bf16.h>

// ---------------------------------------------------------------------------
// Encode-Process-Decode GNN for MI455X (gfx1250, wave32, WMMA).
// All 128x128 dense layers run as fused 4-layer (or 3-layer) WMMA kernels:
//   v_wmma_f32_16x16x32_bf16, weights pre-swizzled to B-fragment layout in LDS
//   (double-buffered ds_load_b128 prefetch, distance 2), activations bounce
//   through per-wave LDS staging between layers (D->A layout transpose).
//   Edge MLP fuses gather(S,R) + atomic scatter(agg) + residual epilogue so
//   the 0.4GB/layer edge activations touch HBM exactly once.
// ---------------------------------------------------------------------------

typedef __attribute__((ext_vector_type(16))) __bf16 v16bf;
typedef __attribute__((ext_vector_type(8)))  float  v8f;

union AFrag { unsigned int u[8]; v16bf v; };
union CFrag { float        f[8]; v8f   v; };

__device__ __forceinline__ unsigned int packbf(float a, float b) {
    union { __bf16 h[2]; unsigned int u; } t;
    t.h[0] = (__bf16)a; t.h[1] = (__bf16)b;
    return t.u;
}

// K-offset inside a 16x32 bf16 A (or 32x16 B) fragment for VGPR v, lane-half h.
__device__ __forceinline__ constexpr int koff(int half, int v) {
    return (v < 4) ? (8 * half + 2 * v) : (16 + 8 * half + 2 * (v - 4));
}

#define NW 8          // waves per block
#define TB 256        // threads per block
#define H  128

// MODE: 0 = bf16 out (opt. relu-last), 1 = f32 out,
//       2 = edge epilogue (gather S/R + atomic agg + residual edge_h io),
//       3 = node epilogue (K0=256 concat input, residual node_h io)
template <int KT0, int NL, int MODE, bool RELU_LAST>
__global__ __launch_bounds__(TB) void mlp_fused(
    int rows,
    const __bf16* __restrict__ inA,      // [rows,128] bf16 activations
    const float*  __restrict__ inB,      // MODE3: concat second half [rows,128] f32
    const float* __restrict__ W0, const float* __restrict__ B0,
    const float* __restrict__ W1, const float* __restrict__ B1,
    const float* __restrict__ W2, const float* __restrict__ B2,
    const float* __restrict__ W3, const float* __restrict__ B3,
    void* outP,                          // MODE0: bf16*, MODE1: float* (may alias inA)
    const long long* __restrict__ senders,
    const long long* __restrict__ receivers,
    const float* __restrict__ Sbuf,
    const float* __restrict__ Rbuf,
    float* __restrict__ agg,
    __bf16* io_resid)                    // MODE2: edge_h io, MODE3: node_h io (alias inA)
{
    constexpr int KTT = KT0 + (NL - 1) * 4;     // total k-tiles over all layers
    constexpr int WD  = KTT * 2048;             // packed-weight dwords in LDS

    extern __shared__ unsigned int smem[];
    unsigned int* wlds = smem;                          // packed bf16 weights
    float*        blds = (float*)(smem + WD);           // biases, NL*128
    __bf16*       stag = (__bf16*)(blds + NL * H);      // NW * 16*128 staging

    // ---- cooperative pack: f32 global weights -> bf16 B-fragment layout ----
    {
        const float* Ws[4] = { W0, W1, W2, W3 };
        const float* Bs[4] = { B0, B1, B2, B3 };
        for (int l = 0; l < NL; ++l) {
            const int kt    = (l == 0) ? KT0 : 4;
            const int baseD = ((l == 0) ? 0 : (KT0 + (l - 1) * 4)) * 2048;
            const float* W  = Ws[l];
            const int tot   = kt * 2048;
            for (int pi = threadIdx.x; pi < tot; pi += blockDim.x) {
                const int blk  = pi >> 8, r = pi & 255;
                const int lane = r >> 3,  v = r & 7;
                const int kb   = blk >> 3, nb = blk & 7;
                const int half = lane >> 4;
                const int n    = nb * 16 + (lane & 15);
                const int k    = kb * 32 + koff(half, v);
                wlds[baseD + pi] = packbf(W[k * H + n], W[(k + 1) * H + n]);
            }
            for (int j = threadIdx.x; j < H; j += blockDim.x)
                blds[l * H + j] = Bs[l][j];
        }
    }
    __syncthreads();

    const int wave = threadIdx.x >> 5;
    const int lane = threadIdx.x & 31;
    const int lm   = lane & 15;        // row (A) / col (B,D) inside tile
    const int lh   = lane >> 4;        // lane half
    __bf16*       myStag  = stag + wave * (16 * H);
    unsigned int* myStagU = (unsigned int*)myStag;

    // B-fragment loader: t = kb*8 + nb within a layer's packed region
    auto ldB = [&](AFrag& f, int baseD, int t) {
        const uint4* bp = (const uint4*)(wlds + baseD + t * 256 + lane * 8);
        *(uint4*)&f.u[0] = bp[0];
        *(uint4*)&f.u[4] = bp[1];
    };

    const int numTiles = (rows + 15) >> 4;

    for (int tile = blockIdx.x * NW + wave; tile < numTiles; tile += gridDim.x * NW) {
        const int rowBase = tile << 4;
        int rowA = rowBase + lm; if (rowA >= rows) rowA = rows - 1;

        // prefetch layer-0 B fragments 0,1 (overlaps with A global loads)
        AFrag bb[2];
        ldB(bb[0], 0, 0);
        ldB(bb[1], 0, 1);

        // MODE2: hoist coalesced int64 index loads under the whole compute
        long long se[8], re[8];
        if constexpr (MODE == 2) {
#pragma unroll
            for (int v = 0; v < 8; ++v) {
                const int m = v + 8 * lh;
                int e = rowBase + m; if (e >= rows) e = rows - 1;
                se[v] = senders[e];
                re[v] = receivers[e];
            }
        }

        // ---- load layer-0 A fragments ----
        AFrag a[KT0];
#pragma unroll
        for (int kb = 0; kb < KT0; ++kb) {
#pragma unroll
            for (int v = 0; v < 8; ++v) {
                const int k = kb * 32 + koff(lh, v);
                if constexpr (KT0 == 8) {
                    if (k >= H) {   // concat half from f32 agg buffer
                        const float* p = inB + (size_t)rowA * H + (k - H);
                        a[kb].u[v] = packbf(p[0], p[1]);
                        continue;
                    }
                }
                a[kb].u[v] = *(const unsigned int*)(inA + (size_t)rowA * H + k);
            }
        }

        CFrag acc[8];
#pragma unroll
        for (int l = 0; l < NL; ++l) {
            const int kt    = (l == 0) ? KT0 : 4;
            const int T     = kt * 8;
            const int baseD = ((l == 0) ? 0 : (KT0 + (l - 1) * 4)) * 2048;
            const int baseDnext = (KT0 + l * 4) * 2048;   // layer l+1 region

#pragma unroll
            for (int nb = 0; nb < 8; ++nb) {
                const float bv = blds[l * H + nb * 16 + lm];
#pragma unroll
                for (int v = 0; v < 8; ++v) acc[nb].f[v] = bv;
            }

            // software-pipelined WMMA stream, B prefetch distance 2
#pragma unroll
            for (int t = 0; t < T; ++t) {
                const int kb = t >> 3, nb = t & 7;
                acc[nb].v = __builtin_amdgcn_wmma_f32_16x16x32_bf16(
                    false, a[kb].v, false, bb[t & 1].v, (short)0, acc[nb].v,
                    false, false);
                if (t + 2 < T) {
                    ldB(bb[t & 1], baseD, t + 2);
                } else if (l < NL - 1) {
                    // prefetch next layer's fragments 0,1 under last WMMAs
                    ldB(bb[t & 1], baseDnext, t + 2 - T);
                }
            }

            if (l < NL - 1) {
                // ReLU -> bf16 -> per-wave LDS staging (D layout) ...
#pragma unroll
                for (int nb = 0; nb < 8; ++nb) {
                    const int n = nb * 16 + lm;
#pragma unroll
                    for (int v = 0; v < 8; ++v) {
                        const int m = v + 8 * lh;
                        float x = acc[nb].f[v];
                        x = x > 0.f ? x : 0.f;
                        myStag[m * H + n] = (__bf16)x;
                    }
                }
                // ... reload as A fragments (compiler inserts s_wait_dscnt)
#pragma unroll
                for (int kb = 0; kb < 4; ++kb)
#pragma unroll
                    for (int v = 0; v < 8; ++v) {
                        const int k = kb * 32 + koff(lh, v);
                        a[kb].u[v] = myStagU[(lm * H + k) >> 1];
                    }
            }
        }

        // ---------------- epilogue ----------------
        if constexpr (MODE == 2) {
#pragma unroll
            for (int nb = 0; nb < 8; ++nb) {
                const int n = nb * 16 + lm;
#pragma unroll
                for (int v = 0; v < 8; ++v) {
                    const int m = v + 8 * lh;
                    const int e = rowBase + m;
                    if (e < rows) {
                        float val = acc[nb].f[v]
                                  + Sbuf[(size_t)se[v] * H + n]
                                  + Rbuf[(size_t)re[v] * H + n];
                        atomicAdd(agg + (size_t)re[v] * H + n, val);
                        const size_t ix = (size_t)e * H + n;
                        const float old = (float)io_resid[ix];
                        io_resid[ix] = (__bf16)(val + old);
                    }
                }
            }
        } else if constexpr (MODE == 3) {
#pragma unroll
            for (int nb = 0; nb < 8; ++nb) {
                const int n = nb * 16 + lm;
#pragma unroll
                for (int v = 0; v < 8; ++v) {
                    const int m = v + 8 * lh;
                    const int row = rowBase + m;
                    if (row < rows) {
                        const size_t ix = (size_t)row * H + n;
                        const float old = (float)io_resid[ix];
                        io_resid[ix] = (__bf16)(acc[nb].f[v] + old);
                    }
                }
            }
        } else if constexpr (MODE == 1) {
            float* out = (float*)outP;
#pragma unroll
            for (int nb = 0; nb < 8; ++nb) {
                const int n = nb * 16 + lm;
#pragma unroll
                for (int v = 0; v < 8; ++v) {
                    const int row = rowBase + v + 8 * lh;
                    if (row < rows) out[(size_t)row * H + n] = acc[nb].f[v];
                }
            }
        } else {
            __bf16* out = (__bf16*)outP;
#pragma unroll
            for (int nb = 0; nb < 8; ++nb) {
                const int n = nb * 16 + lm;
#pragma unroll
                for (int v = 0; v < 8; ++v) {
                    const int row = rowBase + v + 8 * lh;
                    if (row < rows) {
                        float x = acc[nb].f[v];
                        if (RELU_LAST) x = x > 0.f ? x : 0.f;
                        out[(size_t)row * H + n] = (__bf16)x;
                    }
                }
            }
        }
    }
}

// Tiny first layer: [rows,kin] f32 @ W[kin,128] + b, ReLU, bf16 out.
__global__ void k_inproj(long long rows, int kin,
                         const float* __restrict__ in,
                         const float* __restrict__ W,
                         const float* __restrict__ B,
                         __bf16* __restrict__ out) {
    const long long t = (long long)blockIdx.x * blockDim.x + threadIdx.x;
    if (t >= rows * H) return;
    const long long row = t >> 7;
    const int j = (int)(t & 127);
    float acc = B[j];
    for (int k = 0; k < kin; ++k)
        acc += in[row * kin + k] * W[k * H + j];
    out[t] = (__bf16)(acc > 0.f ? acc : 0.f);
}

// Decoder head: [rows,128] bf16 @ W[128,2] + b -> f32 output.
__global__ void k_dec(int rows, const __bf16* __restrict__ h,
                      const float* __restrict__ W,
                      const float* __restrict__ B,
                      float* __restrict__ out) {
    const int t = blockIdx.x * blockDim.x + threadIdx.x;
    if (t >= rows * 2) return;
    const int row = t >> 1, d = t & 1;
    float acc = B[d];
    for (int k = 0; k < H; ++k)
        acc += (float)h[(size_t)row * H + k] * W[k * 2 + d];
    out[t] = acc;
}

static inline size_t smemBytes(int kt0, int nl) {
    return (size_t)(kt0 + (nl - 1) * 4) * 2048 * 4   // packed bf16 weights
         + (size_t)nl * H * 4                         // biases
         + (size_t)NW * 16 * H * 2;                   // per-wave staging
}

extern "C" void kernel_launch(void* const* d_in, const int* in_sizes, int n_in,
                              void* d_out, int out_size, void* d_ws, size_t ws_size,
                              hipStream_t stream) {
    const int N = in_sizes[0] / 3;     // nodes
    const int E = in_sizes[1] / 3;     // edges

    const float*     x   = (const float*)d_in[0];
    const float*     ea  = (const float*)d_in[1];
    const long long* ei  = (const long long*)d_in[2];   // int64 in reference
    const long long* senders   = ei;
    const long long* receivers = ei + E;
    auto Wp = [&](int i) { return (const float*)d_in[i]; };

    // ---- workspace carve (all sizes 256B-aligned) ----
    char* w = (char*)d_ws;
    __bf16* node_h = (__bf16*)w;            w += (size_t)N * H * 2;
    __bf16* edge_h = (__bf16*)w;            w += (size_t)E * H * 2;
    float*  S      = (float*)w;             w += (size_t)N * H * 4;
    float*  R      = (float*)w;             w += (size_t)N * H * 4;
    float*  agg    = (float*)w;             w += (size_t)N * H * 4;
    __bf16* dec_h  = (__bf16*)w;            w += (size_t)N * H * 2;
    (void)ws_size; (void)n_in;

    auto blocksFor = [](int rows) { return (((rows + 15) >> 4) + NW - 1) / NW; };
    const size_t sm3  = smemBytes(4, 3);   // 3x128 dense layers
    const size_t sm4  = smemBytes(4, 4);   // 4x128 dense layers
    const size_t sm4c = smemBytes(8, 4);   // concat-256 first layer

    // ---------------- encoder ----------------
    {
        long long tot = (long long)N * H;
        k_inproj<<<(unsigned)((tot + TB - 1) / TB), TB, 0, stream>>>(N, 3, x, Wp(3), Wp(4), node_h);
        mlp_fused<4, 3, 0, false><<<blocksFor(N), TB, sm3, stream>>>(
            N, node_h, nullptr, Wp(5), Wp(6), Wp(7), Wp(8), Wp(9), Wp(10), nullptr, nullptr,
            node_h, nullptr, nullptr, nullptr, nullptr, nullptr, nullptr);
        tot = (long long)E * H;
        k_inproj<<<(unsigned)((tot + TB - 1) / TB), TB, 0, stream>>>(E, 3, ea, Wp(11), Wp(12), edge_h);
        mlp_fused<4, 3, 0, false><<<blocksFor(E), TB, sm3, stream>>>(
            E, edge_h, nullptr, Wp(13), Wp(14), Wp(15), Wp(16), Wp(17), Wp(18), nullptr, nullptr,
            edge_h, nullptr, nullptr, nullptr, nullptr, nullptr, nullptr);
    }

    // ---------------- processor: 2 GN blocks ----------------
    for (int b = 0; b < 2; ++b) {
        const int p = 19 + b * 32;   // sender, receiver, edge, node (8 leaves each)
        mlp_fused<4, 4, 1, false><<<blocksFor(N), TB, sm4, stream>>>(
            N, node_h, nullptr, Wp(p + 0), Wp(p + 1), Wp(p + 2), Wp(p + 3),
            Wp(p + 4), Wp(p + 5), Wp(p + 6), Wp(p + 7),
            S, nullptr, nullptr, nullptr, nullptr, nullptr, nullptr);
        mlp_fused<4, 4, 1, false><<<blocksFor(N), TB, sm4, stream>>>(
            N, node_h, nullptr, Wp(p + 8), Wp(p + 9), Wp(p + 10), Wp(p + 11),
            Wp(p + 12), Wp(p + 13), Wp(p + 14), Wp(p + 15),
            R, nullptr, nullptr, nullptr, nullptr, nullptr, nullptr);
        hipMemsetAsync(agg, 0, (size_t)N * H * 4, stream);
        // edge MLP + gather(S,R) + atomic scatter(agg) + residual edge update
        mlp_fused<4, 4, 2, false><<<blocksFor(E), TB, sm4, stream>>>(
            E, edge_h, nullptr, Wp(p + 16), Wp(p + 17), Wp(p + 18), Wp(p + 19),
            Wp(p + 20), Wp(p + 21), Wp(p + 22), Wp(p + 23),
            nullptr, senders, receivers, S, R, agg, edge_h);
        // node MLP on concat(node_h, agg) + residual node update
        mlp_fused<8, 4, 3, false><<<blocksFor(N), TB, sm4c, stream>>>(
            N, node_h, agg, Wp(p + 24), Wp(p + 25), Wp(p + 26), Wp(p + 27),
            Wp(p + 28), Wp(p + 29), Wp(p + 30), Wp(p + 31),
            nullptr, nullptr, nullptr, nullptr, nullptr, nullptr, node_h);
    }

    // ---------------- decoder ----------------
    mlp_fused<4, 3, 0, true><<<blocksFor(N), TB, sm3, stream>>>(
        N, node_h, nullptr, Wp(83), Wp(84), Wp(85), Wp(86), Wp(87), Wp(88), nullptr, nullptr,
        dec_h, nullptr, nullptr, nullptr, nullptr, nullptr, nullptr);
    k_dec<<<(N * 2 + TB - 1) / TB, TB, 0, stream>>>(N, dec_h, Wp(89), Wp(90), (float*)d_out);
    (void)out_size;
}
